// GroupedQueryAttention_86285892976745
// MI455X (gfx1250) — matrix operand order repaired and checked by reading, not compiled, not run on hardware
//
#include <hip/hip_runtime.h>

// ---------------------------------------------------------------------------
// GQA forward for MI455X (gfx1250).
//   * all GEMMs: v_wmma_f32_16x16x32_bf16, operands pre-converted to bf16
//   * tile staging: global_load_async_to_lds_b128 (ASYNCcnt), double-buffered
// ---------------------------------------------------------------------------

typedef __attribute__((ext_vector_type(16))) __bf16 v16bf;
typedef __attribute__((ext_vector_type(8)))  float  v8f;
typedef __attribute__((ext_vector_type(4)))  unsigned int u32x4;

#define BM 128
#define BN 128
#define BK 32
#define NTHREADS 256

#define S_LEN   2048
#define D_MODEL 2048
#define N_Q     8
#define N_KV    2
#define HEAD_D  256
#define KV_D    (N_KV * HEAD_D)          // 512
#define ATTN_SCALE 0.0625f               // 256^-0.5

// LDS tiles: A as [m][k], B as [n][k]; bf16 rows of 64B, k-contiguous.
struct SmemTiles {
    unsigned short a[BM][BK] __attribute__((aligned(16)));   // 8 KB
    unsigned short b[BN][BK] __attribute__((aligned(16)));   // 8 KB
};

__device__ __forceinline__ unsigned short f2bf(float f) {
    unsigned int u = __float_as_uint(f);
    unsigned int r = 0x7FFFu + ((u >> 16) & 1u);   // round-to-nearest-even
    return (unsigned short)((u + r) >> 16);
}

// generic pointer -> LDS byte offset (workgroup-relative)
__device__ __forceinline__ unsigned lds_off_of(void* p) {
    return (unsigned)(unsigned long long)
           (__attribute__((address_space(3))) char*)((char*)p);
}

// 16B global -> LDS async copy (no VGPR data path, tracked by ASYNCcnt)
__device__ __forceinline__ void async_cp16(unsigned lds_off,
                                           const unsigned short* gptr) {
    asm volatile("global_load_async_to_lds_b128 %0, %1, off"
                 :: "v"(lds_off),
                    "v"((unsigned long long)(const void*)gptr)
                 : "memory");
}

__device__ __forceinline__ void wait_async0() {
    asm volatile("s_wait_asynccnt 0" ::: "memory");
}

// Stage one 128x32 A tile and 128x32 B tile (bf16) into LDS buffer `buf`.
// 1024 x 16B chunks total, 4 per thread (2 A + 2 B).
__device__ __forceinline__ void stage_tiles(unsigned smA, unsigned smB,
                                            const unsigned short* __restrict__ A, int lda,
                                            const unsigned short* __restrict__ Bt, int ldb,
                                            int m0, int n0, int k0, int tid) {
#pragma unroll
    for (int it = 0; it < 2; ++it) {
        int c   = tid + it * NTHREADS;   // 0..511
        int row = c >> 2;                // 128 rows
        int kc  = c & 3;                 // 16B unit within 64B row
        async_cp16(smA + (unsigned)(row * 64 + kc * 16),
                   A + (size_t)(m0 + row) * lda + k0 + kc * 8);
    }
#pragma unroll
    for (int it = 0; it < 2; ++it) {
        int c   = tid + it * NTHREADS;
        int row = c >> 2;
        int kc  = c & 3;
        async_cp16(smB + (unsigned)(row * 64 + kc * 16),
                   Bt + (size_t)(n0 + row) * ldb + k0 + kc * 8);
    }
}

// A-fragment (16x32 bf16): lane m in [0,16): elems 0..7 = K khalf..khalf+7,
// elems 8..15 = K 16+khalf..; khalf = 8*(lane>=16).
__device__ __forceinline__ v16bf load_a_frag(const unsigned short* base) {
    int lane  = threadIdx.x & 31;
    int row   = lane & 15;
    int khalf = (lane >> 4) << 3;
    const unsigned short* p = base + row * BK;
    union { u32x4 u[2]; v16bf v; } t;
    t.u[0] = *(const u32x4*)(p + khalf);
    t.u[1] = *(const u32x4*)(p + 16 + khalf);
    return t.v;
}

// B-fragment (32x16 bf16): lane n in [0,16): elems 0..7 = K kbase..kbase+7,
// elems 8..15 = K 16+kbase..; kbase = 8*(lane>=16).
__device__ __forceinline__ v16bf load_b_frag(const unsigned short* base) {
    int lane  = threadIdx.x & 31;
    int col   = lane & 15;
    int kbase = (lane >> 4) << 3;
    const unsigned short* p = base + col * BK + kbase;
    union { u32x4 u[2]; v16bf v; } t;
    t.u[0] = *(const u32x4*)(p);
    t.u[1] = *(const u32x4*)(p + 16);
    return t.v;
}

// Epilogue modes
#define EPI_F32    0   // float out, alpha * acc + bias
#define EPI_BF16   1   // bf16 out,  acc + bias
#define EPI_BF16_T 2   // bf16 out transposed: C[n][m] = acc + bias[n]

// C[m0:+128, n0:+128] = A[M][K] @ Bt[N][K]^T. bf16 in, 8 waves, 32x64/wave.
template <int EPI>
__device__ __forceinline__ void gemm_bt(const unsigned short* __restrict__ A, int lda,
                                        const unsigned short* __restrict__ Bt, int ldb,
                                        void* __restrict__ C, int ldc, int K,
                                        int m0, int n0,
                                        const float* __restrict__ bias, float alpha) {
    __shared__ SmemTiles sm[2];
    const unsigned smA[2] = { lds_off_of(&sm[0].a[0][0]), lds_off_of(&sm[1].a[0][0]) };
    const unsigned smB[2] = { lds_off_of(&sm[0].b[0][0]), lds_off_of(&sm[1].b[0][0]) };

    int tid = threadIdx.x;
    int w = tid >> 5;
    int waveM = w >> 1;          // 0..3 -> 32-row slice
    int waveN = w & 1;           // 0..1 -> 64-col slice

    v8f acc[2][4];
#pragma unroll
    for (int i = 0; i < 2; ++i)
#pragma unroll
        for (int j = 0; j < 4; ++j)
#pragma unroll
            for (int r = 0; r < 8; ++r) acc[i][j][r] = 0.0f;

    const int nk = K / BK;
    stage_tiles(smA[0], smB[0], A, lda, Bt, ldb, m0, n0, 0, tid);

    for (int i = 0; i < nk; ++i) {
        wait_async0();           // current buffer's async copies done (this wave)
        __syncthreads();         // all waves done staging + done reading other buf

        if (i + 1 < nk) {        // prefetch next K-tile into the other buffer
            stage_tiles(smA[(i + 1) & 1], smB[(i + 1) & 1],
                        A, lda, Bt, ldb, m0, n0, (i + 1) * BK, tid);
        }

        const unsigned short (*As)[BK] = sm[i & 1].a;
        const unsigned short (*Bs)[BK] = sm[i & 1].b;

        v16bf af[2], bf[4];
#pragma unroll
        for (int tm = 0; tm < 2; ++tm)
            af[tm] = load_a_frag(&As[waveM * 32 + tm * 16][0]);
#pragma unroll
        for (int tn = 0; tn < 4; ++tn)
            bf[tn] = load_b_frag(&Bs[waveN * 64 + tn * 16][0]);

#pragma unroll
        for (int tm = 0; tm < 2; ++tm)
#pragma unroll
            for (int tn = 0; tn < 4; ++tn)
                acc[tm][tn] = __builtin_amdgcn_wmma_f32_16x16x32_bf16(
                    false, af[tm], false, bf[tn], (short)0, acc[tm][tn], false, false);
    }

    // C/D layout: lane n = lane&15, mhalf = 8*(lane>=16), VGPR r -> row mhalf+r
    int lane  = tid & 31;
    int n     = lane & 15;
    int mhalf = (lane >> 4) << 3;
#pragma unroll
    for (int tm = 0; tm < 2; ++tm) {
#pragma unroll
        for (int tn = 0; tn < 4; ++tn) {
            int gm = m0 + waveM * 32 + tm * 16 + mhalf;
            int gn = n0 + waveN * 64 + tn * 16 + n;
            float bv = bias ? bias[gn] : 0.0f;
#pragma unroll
            for (int r = 0; r < 8; ++r) {
                float v = acc[tm][tn][r] * alpha + bv;
                if (EPI == EPI_F32) {
                    ((float*)C)[(size_t)(gm + r) * ldc + gn] = v;
                } else if (EPI == EPI_BF16) {
                    ((unsigned short*)C)[(size_t)(gm + r) * ldc + gn] = f2bf(v);
                } else {
                    ((unsigned short*)C)[(size_t)gn * ldc + gm + r] = f2bf(v);
                }
            }
        }
    }
}

// ------------------------------- kernels -----------------------------------

// y[i] = bf16(x[i]), 4 elements / thread
__global__ __launch_bounds__(NTHREADS)
void gqa_cvt_bf16(const float* __restrict__ in, unsigned short* __restrict__ outp) {
    size_t i = ((size_t)blockIdx.x * NTHREADS + threadIdx.x) * 4;
    float4 v = *(const float4*)(in + i);
    union { unsigned short s[4]; unsigned long long u; } t;
    t.s[0] = f2bf(v.x); t.s[1] = f2bf(v.y); t.s[2] = f2bf(v.z); t.s[3] = f2bf(v.w);
    *(unsigned long long*)(outp + i) = t.u;
}

// Wt[n][k] = bf16(W[k][n]); W is [Kd][Nd] row-major. 32x32 LDS transpose.
__global__ __launch_bounds__(NTHREADS)
void gqa_cvt_bf16_t(const float* __restrict__ W, unsigned short* __restrict__ Wt,
                    int Kd, int Nd) {
    __shared__ float t[32][33];
    int n0 = blockIdx.x * 32, k0 = blockIdx.y * 32;
    int tx = threadIdx.x & 31, ty = threadIdx.x >> 5;    // ty: 0..7
#pragma unroll
    for (int r = 0; r < 32; r += 8)
        t[ty + r][tx] = W[(size_t)(k0 + ty + r) * Nd + n0 + tx];
    __syncthreads();
#pragma unroll
    for (int r = 0; r < 32; r += 8)
        Wt[(size_t)(n0 + ty + r) * Kd + k0 + tx] = f2bf(t[tx][ty + r]);
}

__global__ __launch_bounds__(NTHREADS)
void gqa_gemm_f32(const unsigned short* __restrict__ A, int lda,
                  const unsigned short* __restrict__ Bt, int ldb,
                  float* __restrict__ C, int ldc, int K,
                  const float* __restrict__ bias) {
    gemm_bt<EPI_F32>(A, lda, Bt, ldb, C, ldc, K,
                     blockIdx.y * BM, blockIdx.x * BN, bias, 1.0f);
}

__global__ __launch_bounds__(NTHREADS)
void gqa_gemm_bf16(const unsigned short* __restrict__ A, int lda,
                   const unsigned short* __restrict__ Bt, int ldb,
                   unsigned short* __restrict__ C, int ldc, int K,
                   const float* __restrict__ bias) {
    gemm_bt<EPI_BF16>(A, lda, Bt, ldb, C, ldc, K,
                      blockIdx.y * BM, blockIdx.x * BN, bias, 1.0f);
}

__global__ __launch_bounds__(NTHREADS)
void gqa_gemm_bf16_t(const unsigned short* __restrict__ A, int lda,
                     const unsigned short* __restrict__ Bt, int ldb,
                     unsigned short* __restrict__ C, int ldc, int K,
                     const float* __restrict__ bias) {
    gemm_bt<EPI_BF16_T>(A, lda, Bt, ldb, C, ldc, K,
                        blockIdx.y * BM, blockIdx.x * BN, bias, 1.0f);
}

// scores[b][h][i][j] = scale * Q[b][i][h]·K[b][j][hk]; skip blocks above diag
__global__ __launch_bounds__(NTHREADS)
void gqa_scores(const unsigned short* __restrict__ Qb,
                const unsigned short* __restrict__ Kb,
                float* __restrict__ probs) {
    int n0 = blockIdx.x * BN;
    int m0 = blockIdx.y * BM;
    if (n0 > m0 + (BM - 1)) return;          // uniform per block
    int bh = blockIdx.z;
    int b  = bh >> 3;
    int h  = bh & 7;
    int hk = h >> 2;
    const unsigned short* A  = Qb + (size_t)b * S_LEN * D_MODEL + (size_t)h * HEAD_D;
    const unsigned short* Bt = Kb + (size_t)b * S_LEN * KV_D + (size_t)hk * HEAD_D;
    float* C = probs + (size_t)bh * S_LEN * S_LEN;
    gemm_bt<EPI_F32>(A, D_MODEL, Bt, KV_D, C, S_LEN, HEAD_D, m0, n0,
                     nullptr, ATTN_SCALE);
}

// O[b][i][h*256+d] = sum_j probs_bf[b][h][i][j] * Vt[hk*256+d][b*2048+j]
__global__ __launch_bounds__(NTHREADS)
void gqa_attn_out(const unsigned short* __restrict__ Pb,
                  const unsigned short* __restrict__ Vt,
                  unsigned short* __restrict__ Ows) {
    int n0 = blockIdx.x * BN;
    int m0 = blockIdx.y * BM;
    int bh = blockIdx.z;
    int b  = bh >> 3;
    int h  = bh & 7;
    int hk = h >> 2;
    const unsigned short* A  = Pb + (size_t)bh * S_LEN * S_LEN;               // lda S
    const unsigned short* Bt = Vt + (size_t)(hk * HEAD_D) * (2 * S_LEN) + (size_t)b * S_LEN;
    unsigned short* C = Ows + (size_t)b * S_LEN * D_MODEL + (size_t)h * HEAD_D;
    gemm_bt<EPI_BF16>(A, S_LEN, Bt, 2 * S_LEN, C, D_MODEL, S_LEN, m0, n0,
                      nullptr, 1.0f);
}

// Row-wise causal softmax in place; dual-writes fp32 (d_out) + bf16 (ws).
__global__ __launch_bounds__(NTHREADS)
void gqa_softmax(float* __restrict__ probs, unsigned short* __restrict__ probs_bf) {
    int r = blockIdx.x;
    int i = r & (S_LEN - 1);
    float* row = probs + (size_t)r * S_LEN;
    unsigned short* rb = probs_bf + (size_t)r * S_LEN;
    int L = i + 1;

    __shared__ float red[NTHREADS / 32];
    int tid = threadIdx.x, lane = tid & 31, wv = tid >> 5;

    float m = -3.0e38f;
    for (int j = tid; j < L; j += NTHREADS) m = fmaxf(m, row[j]);
#pragma unroll
    for (int off = 16; off; off >>= 1) m = fmaxf(m, __shfl_xor(m, off, 32));
    if (lane == 0) red[wv] = m;
    __syncthreads();
    if (wv == 0) {
        float t = (lane < NTHREADS / 32) ? red[lane] : -3.0e38f;
#pragma unroll
        for (int off = 4; off; off >>= 1) t = fmaxf(t, __shfl_xor(t, off, 32));
        if (lane == 0) red[0] = t;
    }
    __syncthreads();
    m = red[0];
    __syncthreads();

    float s = 0.0f;
    for (int j = tid; j < L; j += NTHREADS) s += __expf(row[j] - m);
#pragma unroll
    for (int off = 16; off; off >>= 1) s += __shfl_xor(s, off, 32);
    if (lane == 0) red[wv] = s;
    __syncthreads();
    if (wv == 0) {
        float t = (lane < NTHREADS / 32) ? red[lane] : 0.0f;
#pragma unroll
        for (int off = 4; off; off >>= 1) t += __shfl_xor(t, off, 32);
        if (lane == 0) red[0] = t;
    }
    __syncthreads();
    float inv = 1.0f / red[0];

    for (int j = tid; j < L; j += NTHREADS) {
        float p = __expf(row[j] - m) * inv;
        row[j] = p;
        rb[j]  = f2bf(p);
    }
    for (int j = L + tid; j < S_LEN; j += NTHREADS) {
        row[j] = 0.0f;
        rb[j]  = 0;
    }
}

// ------------------------------- launcher ----------------------------------

extern "C" void kernel_launch(void* const* d_in, const int* in_sizes, int n_in,
                              void* d_out, int out_size, void* d_ws, size_t ws_size,
                              hipStream_t stream) {
    const float* x  = (const float*)d_in[0];
    const float* Wq = (const float*)d_in[1];
    const float* bq = (const float*)d_in[2];
    const float* Wk = (const float*)d_in[3];
    const float* bk = (const float*)d_in[4];
    const float* Wv = (const float*)d_in[5];
    const float* bv = (const float*)d_in[6];
    const float* Wo = (const float*)d_in[7];
    const float* bo = (const float*)d_in[8];

    const int B = 2;
    const int M = B * S_LEN;                 // 4096
    float* out   = (float*)d_out;
    float* probs = (float*)d_out + (size_t)M * D_MODEL;

    // bf16 workspace
    unsigned short* xb   = (unsigned short*)d_ws;                    // [4096][2048]
    unsigned short* Wqt  = xb   + (size_t)M * D_MODEL;               // [2048][2048]
    unsigned short* Wkt  = Wqt  + (size_t)D_MODEL * D_MODEL;         // [512][2048]
    unsigned short* Wvt  = Wkt  + (size_t)KV_D * D_MODEL;            // [512][2048]
    unsigned short* Wot  = Wvt  + (size_t)KV_D * D_MODEL;            // [2048][2048]
    unsigned short* Qb   = Wot  + (size_t)D_MODEL * D_MODEL;         // [4096][2048]
    unsigned short* Kb   = Qb   + (size_t)M * D_MODEL;               // [4096][512]
    unsigned short* Vt   = Kb   + (size_t)M * KV_D;                  // [512][4096]
    unsigned short* Ows  = Vt   + (size_t)KV_D * M;                  // [4096][2048]
    unsigned short* Pbf  = Ows  + (size_t)M * D_MODEL;               // [16][2048][2048]

    dim3 blk(NTHREADS);

    // fp32 -> bf16 conversions (x direct; weights transposed to [n][k])
    gqa_cvt_bf16<<<dim3(((size_t)M * D_MODEL) / (NTHREADS * 4)), blk, 0, stream>>>(x, xb);
    gqa_cvt_bf16_t<<<dim3(D_MODEL / 32, D_MODEL / 32), blk, 0, stream>>>(Wq, Wqt, D_MODEL, D_MODEL);
    gqa_cvt_bf16_t<<<dim3(KV_D / 32,    D_MODEL / 32), blk, 0, stream>>>(Wk, Wkt, D_MODEL, KV_D);
    gqa_cvt_bf16_t<<<dim3(KV_D / 32,    D_MODEL / 32), blk, 0, stream>>>(Wv, Wvt, D_MODEL, KV_D);
    gqa_cvt_bf16_t<<<dim3(D_MODEL / 32, D_MODEL / 32), blk, 0, stream>>>(Wo, Wot, D_MODEL, D_MODEL);

    // projections (bf16 outputs; V written transposed as Vt[d][row])
    gqa_gemm_bf16<<<dim3(D_MODEL / BN, M / BM), blk, 0, stream>>>(
        xb, D_MODEL, Wqt, D_MODEL, Qb, D_MODEL, D_MODEL, bq);
    gqa_gemm_bf16<<<dim3(KV_D / BN, M / BM), blk, 0, stream>>>(
        xb, D_MODEL, Wkt, D_MODEL, Kb, KV_D, D_MODEL, bk);
    gqa_gemm_bf16_t<<<dim3(KV_D / BN, M / BM), blk, 0, stream>>>(
        xb, D_MODEL, Wvt, D_MODEL, Vt, M, D_MODEL, bv);

    // raw scaled scores into fp32 probs buffer (upper-triangle blocks skipped)
    gqa_scores<<<dim3(S_LEN / BN, S_LEN / BM, B * N_Q), blk, 0, stream>>>(Qb, Kb, probs);

    // causal softmax: fp32 in place + bf16 copy
    gqa_softmax<<<dim3(B * N_Q * S_LEN), blk, 0, stream>>>(probs, Pbf);

    // O = probs @ V
    gqa_attn_out<<<dim3(HEAD_D / BN, S_LEN / BM, B * N_Q), blk, 0, stream>>>(Pbf, Vt, Ows);

    // out = O @ Wo + bo
    gqa_gemm_f32<<<dim3(D_MODEL / BN, M / BM), blk, 0, stream>>>(
        Ows, D_MODEL, Wot, D_MODEL, out, D_MODEL, D_MODEL, bo);
}
